// MultiHeadAttention_652835029555
// MI455X (gfx1250) — matrix operand order repaired and checked by reading, not compiled, not run on hardware
//
#include <hip/hip_runtime.h>
#include <hip/hip_bf16.h>

typedef __attribute__((ext_vector_type(16))) _Float16 v16h;
typedef __attribute__((ext_vector_type(8)))  _Float16 v8h;
typedef __attribute__((ext_vector_type(8)))  float    v8f;

#define BB 16
#define LL 512
#define CH 1024   // IN_CH == OUT_CH == HID
#define NH 16
#define DHD 64
#define MROWS (BB*LL)   // 8192

// ---------------------------------------------------------------------------
// gfx1250 async copy: per-lane 16B global -> LDS, tracked by ASYNCcnt
// ---------------------------------------------------------------------------
__device__ __forceinline__ void async_b128(_Float16* lds_dst, const _Float16* gsrc) {
  unsigned loff = (unsigned)(size_t)lds_dst;               // LDS aperture: addr[31:0]
  unsigned long long ga = (unsigned long long)(size_t)gsrc;
  asm volatile("global_load_async_to_lds_b128 %0, %1, off"
               :: "v"(loff), "v"(ga) : "memory");
}
__device__ __forceinline__ void wait_asynccnt0() {
  asm volatile("s_wait_asynccnt 0x0" ::: "memory");
}

// ---------------------------------------------------------------------------
// f32 -> f16 conversion, 8 elements per thread
// ---------------------------------------------------------------------------
__global__ __launch_bounds__(256) void cvt_f32_f16_kernel(
    const float* __restrict__ in, _Float16* __restrict__ out, int n8)
{
  int i = blockIdx.x * blockDim.x + threadIdx.x;
  if (i >= n8) return;
  const float4* p = (const float4*)(in + i * 8);
  float4 a = p[0], b = p[1];
  v8h o;
  o[0]=(_Float16)a.x; o[1]=(_Float16)a.y; o[2]=(_Float16)a.z; o[3]=(_Float16)a.w;
  o[4]=(_Float16)b.x; o[5]=(_Float16)b.y; o[6]=(_Float16)b.z; o[7]=(_Float16)b.w;
  *(v8h*)(out + i * 8) = o;
}

// ---------------------------------------------------------------------------
// GEMM: Y = A[M,K] @ W[N,K]^T + bias.  One wave computes a 32x64 strip
// (8 WMMAs per 32-deep k-step, B fragments reused across two A rows).
// mode 0: store f16 into [B,H,L,DH]   (Q, K)
// mode 1: store f16 into [B,H,DH,L]   (V transposed)
// mode 2: store f32 into [M, CH]      (final projection -> d_out)
// ---------------------------------------------------------------------------
__global__ __launch_bounds__(256) void gemm_wmma_kernel(
    const _Float16* __restrict__ A,
    const _Float16* __restrict__ W,
    const float* __restrict__ bias,
    void* __restrict__ outp,
    int mode)
{
  const int wave   = blockIdx.x * (blockDim.x >> 5) + (threadIdx.x >> 5);
  const int lane   = threadIdx.x & 31;
  const int mtile  = wave >> 4;          // 0..255  (M/32)
  const int nstrip = wave & 15;          // 0..15   (CH/64)
  const int hl = lane >> 4;              // lane half
  const int lm = lane & 15;
  const int kb = hl * 8;
  const int koff = hl * 16;

  v8f acc[2][4] = {};
  const _Float16* arow0 = A + (size_t)(mtile * 32 + lm) * CH;
  const _Float16* arow1 = A + (size_t)(mtile * 32 + 16 + lm) * CH;

  for (int k0 = 0; k0 < CH; k0 += 32) {
    v16h af0, af1;
    {
      v8h lo = *(const v8h*)(arow0 + k0 + koff);
      v8h hi = *(const v8h*)(arow0 + k0 + koff + 8);
#pragma unroll
      for (int i = 0; i < 8; ++i) { af0[i] = lo[i]; af0[8 + i] = hi[i]; }
    }
    {
      v8h lo = *(const v8h*)(arow1 + k0 + koff);
      v8h hi = *(const v8h*)(arow1 + k0 + koff + 8);
#pragma unroll
      for (int i = 0; i < 8; ++i) { af1[i] = lo[i]; af1[8 + i] = hi[i]; }
    }
#pragma unroll
    for (int s = 0; s < 4; ++s) {
      const _Float16* wp =
          W + (size_t)(nstrip * 64 + s * 16 + lm) * CH + k0 + koff;
      v16h bf = *(const v16h*)wp;
      acc[0][s] = __builtin_amdgcn_wmma_f32_16x16x32_f16(
          false, af0, false, bf, (short)0, acc[0][s], false, false);
      acc[1][s] = __builtin_amdgcn_wmma_f32_16x16x32_f16(
          false, af1, false, bf, (short)0, acc[1][s], false, false);
    }
  }

#pragma unroll
  for (int g = 0; g < 2; ++g) {
#pragma unroll
    for (int s = 0; s < 4; ++s) {
      const int col = nstrip * 64 + s * 16 + lm;
      const float bv = bias[col];
#pragma unroll
      for (int r = 0; r < 8; ++r) {
        const float val = acc[g][s][r] + bv;
        const int grow = mtile * 32 + g * 16 + r + 8 * hl;
        if (mode == 2) {
          ((float*)outp)[(size_t)grow * CH + col] = val;
        } else {
          const int b = grow >> 9, l = grow & 511;
          const int h = col >> 6, d = col & 63;
          _Float16* o = (_Float16*)outp;
          if (mode == 0)
            o[(((size_t)b * NH + h) * LL + l) * DHD + d] = (_Float16)val;
          else
            o[(((size_t)b * NH + h) * DHD + d) * LL + l] = (_Float16)val;
        }
      }
    }
  }
}

// ---------------------------------------------------------------------------
// Flash attention. One block = one (b,h,qgroup) triple, 8 waves = 8 q-tiles.
// K/V tiles are staged cooperatively into LDS with async-to-LDS DMA
// (double buffered, ASYNCcnt + barrier fenced); P->A transpose via LDS.
// Mask replicates reference broadcast quirk: dead iff
//   seq_mask[h,q]*seq_mask[h,k] + (-k * 2^(-h/2)) == 0
// ---------------------------------------------------------------------------
__global__ __launch_bounds__(256) void attn_wmma_kernel(
    const _Float16* __restrict__ Q,    // [B,H,L,DH]
    const _Float16* __restrict__ K,    // [B,H,L,DH]
    const _Float16* __restrict__ Vt,   // [B,H,DH,L]
    const float* __restrict__ seq_mask,// [B,L]
    _Float16* __restrict__ O)          // [B,L,CH]
{
  __shared__ _Float16 kts[2][32 * DHD];   // [kpos][d]
  __shared__ _Float16 vts[2][DHD * 32];   // [d][kpos]
  __shared__ _Float16 plds[8][16 * 32];

  const int tid  = threadIdx.x;
  const int wid  = tid >> 5;
  const int lane = tid & 31;
  const int qg = blockIdx.x & 3;          // q-tile group
  const int bh = blockIdx.x >> 2;         // 0..255
  const int h = bh & 15;
  const int b = bh >> 4;
  const int qt = qg * 8 + wid;            // this wave's q-tile (0..31)
  const int hl = lane >> 4, lm = lane & 15;
  const int kb = hl * 8, koff = hl * 16;

  const _Float16* Qb = Q + ((size_t)bh * LL + qt * 16) * DHD;
  const _Float16* Kb = K + (size_t)bh * LL * DHD;
  const _Float16* Vb = Vt + (size_t)bh * DHD * LL;

  // cooperative tile-fill mapping: one b128 per thread per tile
  const int krow = tid >> 3, kch = tid & 7;   // K tile: 32 rows x 64 halves
  const int vrow = tid >> 2, vch = tid & 3;   // V tile: 64 rows x 32 halves

  // Q fragments for d-chunks [0,32) and [32,64)
  v16h qf[2];
#pragma unroll
  for (int c = 0; c < 2; ++c) {
    v8h lo = *(const v8h*)(Qb + lm * DHD + c * 32 + koff);
    v8h hi = *(const v8h*)(Qb + lm * DHD + c * 32 + koff + 8);
#pragma unroll
    for (int i = 0; i < 8; ++i) { qf[c][i] = lo[i]; qf[c][8 + i] = hi[i]; }
  }

  const float slope = exp2f(-0.5f * (float)h);
  float smq[8], rm[8], rs[8];
#pragma unroll
  for (int r = 0; r < 8; ++r) {
    rm[r] = -__builtin_inff();
    rs[r] = 0.0f;
    smq[r] = seq_mask[h * LL + qt * 16 + r + 8 * hl];
  }
  v8f accO[4] = {};
  _Float16* pl = plds[wid];

  // prologue: async-fill buffer 0 with tiles at kc = 0
  async_b128(&kts[0][krow * DHD + kch * 8], Kb + (size_t)krow * DHD + kch * 8);
  async_b128(&vts[0][vrow * 32 + vch * 8], Vb + (size_t)vrow * LL + vch * 8);
  wait_asynccnt0();
  __syncthreads();

  for (int it = 0; it < LL / 32; ++it) {
    const int kc = it * 32;
    const int buf = it & 1;
    // prefetch next k-chunk into the other buffer (readers of it finished
    // before the barrier that ended the previous iteration)
    if (it + 1 < LL / 32) {
      const int kn = kc + 32;
      async_b128(&kts[buf ^ 1][krow * DHD + kch * 8],
                 Kb + (size_t)(kn + krow) * DHD + kch * 8);
      async_b128(&vts[buf ^ 1][vrow * 32 + vch * 8],
                 Vb + (size_t)vrow * LL + kn + vch * 8);
    }
    const _Float16* ktb = kts[buf];
    const _Float16* vtb = vts[buf];

    // ---- scores: two 16x16 kpos tiles, 2 WMMAs each over d=64 (from LDS)
    v8f sc[2];
#pragma unroll
    for (int t = 0; t < 2; ++t) {
      v16h kf0 = *(const v16h*)(ktb + (t * 16 + lm) * DHD + koff);
      v16h kf1 = *(const v16h*)(ktb + (t * 16 + lm) * DHD + 32 + koff);
      v8f z = {};
      z = __builtin_amdgcn_wmma_f32_16x16x32_f16(false, qf[0], false, kf0,
                                                 (short)0, z, false, false);
      z = __builtin_amdgcn_wmma_f32_16x16x32_f16(false, qf[1], false, kf1,
                                                 (short)0, z, false, false);
      sc[t] = z;
    }
    // ---- scale + mask (per-lane column kpos is fixed per tile)
#pragma unroll
    for (int t = 0; t < 2; ++t) {
      const int kpos = kc + t * 16 + lm;
      const float smk = seq_mask[h * LL + kpos];
      const float ab = (-(float)kpos) * slope;
#pragma unroll
      for (int r = 0; r < 8; ++r) {
        const float mv = smq[r] * smk + ab;
        const float v = sc[t][r] * 0.125f;   // 1/sqrt(64)
        sc[t][r] = (mv == 0.0f) ? -__builtin_inff() : v;
      }
    }
    // ---- online softmax update (rows live across 16-lane groups)
#pragma unroll
    for (int r = 0; r < 8; ++r) {
      float cmax = fmaxf(sc[0][r], sc[1][r]);
#pragma unroll
      for (int off = 1; off < 16; off <<= 1)
        cmax = fmaxf(cmax, __shfl_xor(cmax, off, 32));
      const float newm = fmaxf(rm[r], cmax);
      const float sold = __expf(rm[r] - newm);
      const float p0 = __expf(sc[0][r] - newm);
      const float p1 = __expf(sc[1][r] - newm);
      float psum = p0 + p1;
#pragma unroll
      for (int off = 1; off < 16; off <<= 1)
        psum += __shfl_xor(psum, off, 32);
      rs[r] = rs[r] * sold + psum;
      rm[r] = newm;
      sc[0][r] = p0; sc[1][r] = p1;
#pragma unroll
      for (int s4 = 0; s4 < 4; ++s4) accO[s4][r] *= sold;
    }
    // ---- P (C layout) -> per-wave LDS -> A layout
#pragma unroll
    for (int t = 0; t < 2; ++t)
#pragma unroll
      for (int r = 0; r < 8; ++r)
        pl[(r + 8 * hl) * 32 + t * 16 + lm] = (_Float16)sc[t][r];
    __asm__ volatile("s_wait_dscnt 0" ::: "memory");
    v16h pa;
    {
      v8h lo = *(const v8h*)(pl + lm * 32 + koff);
      v8h hi = *(const v8h*)(pl + lm * 32 + koff + 8);
#pragma unroll
      for (int i = 0; i < 8; ++i) { pa[i] = lo[i]; pa[8 + i] = hi[i]; }
    }
    __asm__ volatile("s_wait_dscnt 0" ::: "memory");
    // ---- O += P @ V  (V tile in LDS, contiguous in kpos)
#pragma unroll
    for (int s4 = 0; s4 < 4; ++s4) {
      v16h vf = *(const v16h*)(vtb + (s4 * 16 + lm) * 32 + koff);
      accO[s4] = __builtin_amdgcn_wmma_f32_16x16x32_f16(
          false, pa, false, vf, (short)0, accO[s4], false, false);
    }
    // own prefetch landed + everyone done reading this buffer
    wait_asynccnt0();
    __syncthreads();
  }

  // ---- epilogue: divide by rowsum, store f16 [B,L,CH]
#pragma unroll
  for (int s4 = 0; s4 < 4; ++s4) {
    const int d = s4 * 16 + lm;
#pragma unroll
    for (int r = 0; r < 8; ++r) {
      const int grow = b * LL + qt * 16 + r + 8 * hl;
      const float val = accO[s4][r] / rs[r];
      O[(size_t)grow * CH + h * DHD + d] = (_Float16)val;
    }
  }
}

// ---------------------------------------------------------------------------
extern "C" void kernel_launch(void* const* d_in, const int* in_sizes, int n_in,
                              void* d_out, int out_size, void* d_ws, size_t ws_size,
                              hipStream_t stream) {
  (void)in_sizes; (void)n_in; (void)out_size; (void)ws_size;
  const float* x   = (const float*)d_in[0];
  const float* sm  = (const float*)d_in[1];
  const float* Wq  = (const float*)d_in[2];
  const float* bq  = (const float*)d_in[3];
  const float* Wk  = (const float*)d_in[4];
  const float* bk  = (const float*)d_in[5];
  const float* Wv  = (const float*)d_in[6];
  const float* bv  = (const float*)d_in[7];
  const float* Wo  = (const float*)d_in[8];
  const float* bo  = (const float*)d_in[9];
  float* out = (float*)d_out;

  char* ws = (char*)d_ws;
  const size_t SZ_X = (size_t)MROWS * CH * 2;   // 16 MB f16
  const size_t SZ_W = (size_t)CH * CH * 2;      //  2 MB f16
  _Float16* xh   = (_Float16*)(ws);
  _Float16* wqh  = (_Float16*)(ws + SZ_X);
  _Float16* wkh  = (_Float16*)(ws + SZ_X + SZ_W);
  _Float16* wvh  = (_Float16*)(ws + SZ_X + 2 * SZ_W);
  _Float16* woh  = (_Float16*)(ws + SZ_X + 3 * SZ_W);
  _Float16* qb   = (_Float16*)(ws + SZ_X + 4 * SZ_W);
  _Float16* kbuf = (_Float16*)(ws + 2 * SZ_X + 4 * SZ_W);
  _Float16* vtb  = (_Float16*)(ws + 3 * SZ_X + 4 * SZ_W);
  _Float16* attn = (_Float16*)(ws + 4 * SZ_X + 4 * SZ_W);

  // 1) f32 -> f16 conversions
  cvt_f32_f16_kernel<<<(MROWS * CH / 8 + 255) / 256, 256, 0, stream>>>(x, xh, MROWS * CH / 8);
  cvt_f32_f16_kernel<<<(CH * CH / 8 + 255) / 256, 256, 0, stream>>>(Wq, wqh, CH * CH / 8);
  cvt_f32_f16_kernel<<<(CH * CH / 8 + 255) / 256, 256, 0, stream>>>(Wk, wkh, CH * CH / 8);
  cvt_f32_f16_kernel<<<(CH * CH / 8 + 255) / 256, 256, 0, stream>>>(Wv, wvh, CH * CH / 8);
  cvt_f32_f16_kernel<<<(CH * CH / 8 + 255) / 256, 256, 0, stream>>>(Wo, woh, CH * CH / 8);

  // 2) QKV projections: 256 mtiles * 16 nstrips = 4096 wave-jobs -> 512 blocks
  const int GEMM_BLOCKS = (MROWS / 32) * (CH / 64) / 8;   // 512
  gemm_wmma_kernel<<<GEMM_BLOCKS, 256, 0, stream>>>(xh, wqh, bq, (void*)qb, 0);
  gemm_wmma_kernel<<<GEMM_BLOCKS, 256, 0, stream>>>(xh, wkh, bk, (void*)kbuf, 0);
  gemm_wmma_kernel<<<GEMM_BLOCKS, 256, 0, stream>>>(xh, wvh, bv, (void*)vtb, 1);

  // 3) flash attention: one block per (b,h,qgroup) = 16*16*4 = 1024 blocks
  attn_wmma_kernel<<<1024, 256, 0, stream>>>(qb, kbuf, vtb, sm, attn);

  // 4) output projection -> f32 d_out
  gemm_wmma_kernel<<<GEMM_BLOCKS, 256, 0, stream>>>(attn, woh, bo, (void*)out, 2);
}
